// Decoder_27633819582553
// MI455X (gfx1250) — matrix operand order repaired and checked by reading, not compiled, not run on hardware
//
#include <hip/hip_runtime.h>

typedef _Float16 h16;
typedef __attribute__((ext_vector_type(16))) _Float16 v16h;
typedef __attribute__((ext_vector_type(8)))  _Float16 v8h;
typedef __attribute__((ext_vector_type(8)))  float    v8f;
typedef __attribute__((ext_vector_type(4)))  int      v4i;

constexpr int kB  = 2;
constexpr int kT  = 2048;
constexpr int kD  = 1024;
constexpr int kF  = 4096;
constexpr int kH  = 16;
constexpr int kDH = 64;
constexpr int kM  = kB * kT;   // 4096 token rows

#define AS1 __attribute__((address_space(1)))
#define AS3 __attribute__((address_space(3)))

// ---------------------------------------------------------------------------
// WMMA fragment helpers (layouts per CDNA5 ISA 7.12.2)
// ---------------------------------------------------------------------------

// A-matrix 16x32 f16: lanes 0-15 row M=lane, K=0..7 (v0-3) & 16..23 (v4-7);
// lanes 16-31 same rows, K=8..15 & 24..31.
__device__ __forceinline__ v16h load_a_frag(const h16* p, int ld) {
  int lane = threadIdx.x & 31;
  int hs = lane >> 4, r = lane & 15;
  const h16* p0 = p + (size_t)r * ld + 8 * hs;
  v8h a0 = *(const v8h*)p0;
  v8h a1 = *(const v8h*)(p0 + 16);
  v16h out;
#pragma unroll
  for (int i = 0; i < 8; ++i) { out[i] = a0[i]; out[8 + i] = a1[i]; }
  return out;
}

// B-matrix 32x16 f16 where B[k][n] = W[n][k] (row-major W, ld = row stride):
// lanes 0-15 hold K=0..7 & 16..23 of column n=lane; lanes 16-31 hold K=8..15 & 24..31.
__device__ __forceinline__ v16h load_b_frag(const h16* p, int ld) {
  int lane = threadIdx.x & 31;
  int n = lane & 15, kh = lane >> 4;
  const h16* p0 = p + (size_t)n * ld + 8 * kh;
  v8h b0 = *(const v8h*)p0;
  v8h b1 = *(const v8h*)(p0 + 16);
  v16h out;
#pragma unroll
  for (int i = 0; i < 8; ++i) { out[i] = b0[i]; out[8 + i] = b1[i]; }
  return out;
}

__device__ __forceinline__ v8f wmma_f16(v16h a, v16h b, v8f c) {
  return __builtin_amdgcn_wmma_f32_16x16x32_f16(false, a, false, b,
                                                (short)0, c, false, false);
}

// ---------------------------------------------------------------------------
// Async global -> LDS copy (16 bytes per lane), ASYNCcnt-tracked.
// Builtin signature (from hipcc diagnostics): (global int4*, lds int4*, int, int)
// ---------------------------------------------------------------------------
#if defined(__has_builtin) && __has_builtin(__builtin_amdgcn_global_load_async_to_lds_b128)
#define HAVE_ASYNC_LDS 1
#else
#define HAVE_ASYNC_LDS 0
#endif

__device__ __forceinline__ void async_cp_b128(const h16* g, h16* l) {
#if HAVE_ASYNC_LDS
  __builtin_amdgcn_global_load_async_to_lds_b128(
      (AS1 v4i*)(h16*)g, (AS3 v4i*)l, 0, 0);
#else
  *(v8h*)l = *(const v8h*)g;        // synchronous fallback
#endif
}

__device__ __forceinline__ void wait_async_le2() {
#if defined(__has_builtin) && __has_builtin(__builtin_amdgcn_s_wait_asynccnt)
  __builtin_amdgcn_s_wait_asynccnt(2);
#else
  asm volatile("s_wait_asynccnt 0x2" ::: "memory");
#endif
}
__device__ __forceinline__ void wait_async_0() {
#if defined(__has_builtin) && __has_builtin(__builtin_amdgcn_s_wait_asynccnt)
  __builtin_amdgcn_s_wait_asynccnt(0);
#else
  asm volatile("s_wait_asynccnt 0x0" ::: "memory");
#endif
}

// ---------------------------------------------------------------------------
// f32 -> f16 conversion
// ---------------------------------------------------------------------------
__global__ void cvt_f32_f16(const float* __restrict__ in, h16* __restrict__ out,
                            int n) {
  int i = blockIdx.x * blockDim.x + threadIdx.x;
  if (i < n) out[i] = (h16)in[i];
}

// ---------------------------------------------------------------------------
// GEMM: out[m, n] = sum_k A[m,k] * W[n,k] + bias[n]
// block = 128 threads (4 waves); block tile 128(M) x 64(N); wave w owns rows
// [blockIdx.y*128 + 32w, +32).  The shared 64x32 B tile is staged in LDS via
// async global->LDS copies (double buffered); each wave reads B fragments
// with ds_load_b128.  8 WMMAs per wave per 32-wide K step.
// mode 0: f16 row-major out (optional ReLU)
// mode 1: f16 out in (B,H,T,dh) layout (Q/K)
// mode 2: f16 out in (B,H,dh,T) layout (V transposed)
// mode 3: f32 out = acc + bias + residual (row-major)
// ---------------------------------------------------------------------------
__global__ __launch_bounds__(128) void gemm_ws(
    const h16* __restrict__ A, const h16* __restrict__ W,
    const float* __restrict__ bias, const float* __restrict__ resid,
    float* __restrict__ outF, h16* __restrict__ outH,
    int M, int N, int K, int mode, int relu) {
  __shared__ __align__(128) h16 Bbuf[2][64 * 32];   // 2 x 4KB, double buffer

  int tid  = threadIdx.x;
  int wave = tid >> 5;
  int lane = tid & 31;
  int mBase = blockIdx.y * 128 + wave * 32;
  int nBase = blockIdx.x * 64;

  v8f acc[2][4];
#pragma unroll
  for (int mi = 0; mi < 2; ++mi)
#pragma unroll
    for (int f = 0; f < 4; ++f) acc[mi][f] = v8f{};

  // Cooperative B-tile loader: 64 rows x 32 halfs; thread t covers row t>>1,
  // half-row (t&1)*16 .. +16 (32 bytes = two b128 copies).
  int brow = tid >> 1, bcol = (tid & 1) * 16;
  const h16* Bgbase = W + (size_t)(nBase + brow) * K + bcol;

  auto issueB = [&](int buf, int k) {
    const h16* g = Bgbase + k;
    h16* l = &Bbuf[buf][brow * 32 + bcol];
    async_cp_b128(g, l);
    async_cp_b128(g + 8, l + 8);
  };

  int nk = K >> 5;
  issueB(0, 0);
  const h16* Arow = A + (size_t)mBase * K;

  for (int kt = 0; kt < nk; ++kt) {
    int buf = kt & 1;
    int k = kt << 5;
    if (kt + 1 < nk) { issueB(buf ^ 1, k + 32); wait_async_le2(); }
    else             { wait_async_0(); }
    __syncthreads();                       // all waves' B tile visible

    v16h a0 = load_a_frag(Arow + k, K);
    v16h a1 = load_a_frag(Arow + (size_t)16 * K + k, K);
    const h16* bl = &Bbuf[buf][0];
#pragma unroll
    for (int f = 0; f < 4; ++f) {
      v16h b = load_b_frag(bl + f * 16 * 32, 32);
      acc[0][f] = wmma_f16(a0, b, acc[0][f]);
      acc[1][f] = wmma_f16(a1, b, acc[1][f]);
    }
    __syncthreads();                       // done reading before overwrite
  }

  int hs = lane >> 4, n16 = lane & 15;
#pragma unroll
  for (int mi = 0; mi < 2; ++mi) {
#pragma unroll
    for (int f = 0; f < 4; ++f) {
#pragma unroll
      for (int j = 0; j < 8; ++j) {
        int mrow = mBase + mi * 16 + j + 8 * hs;
        int col  = nBase + f * 16 + n16;
        float v = acc[mi][f][j] + bias[col];
        if (mode == 0) {
          if (relu) v = fmaxf(v, 0.f);
          outH[(size_t)mrow * N + col] = (h16)v;
        } else if (mode == 1) {
          int b_ = mrow / kT, t = mrow % kT;
          int hh = col >> 6, d = col & 63;
          outH[((size_t)(b_ * kH + hh) * kT + t) * kDH + d] = (h16)v;
        } else if (mode == 2) {
          int b_ = mrow / kT, t = mrow % kT;
          int hh = col >> 6, d = col & 63;
          outH[((size_t)(b_ * kH + hh) * kDH + d) * kT + t] = (h16)v;
        } else {
          size_t idx = (size_t)mrow * N + col;
          outF[idx] = v + resid[idx];
        }
      }
    }
  }
}

// ---------------------------------------------------------------------------
// Flash-style causal attention, one wave per 16-row query tile.
// qs, ks: (B,H,T,dh) f16.  vt: (B,H,dh,T) f16.  ctx: (B,T,D) f16.
// Online softmax; P tile staged through LDS to become an A-fragment.
// ---------------------------------------------------------------------------
__global__ __launch_bounds__(128) void attn_kernel(
    const h16* __restrict__ qs, const h16* __restrict__ ks,
    const h16* __restrict__ vt, h16* __restrict__ ctx) {
  __shared__ __align__(64) _Float16 Pb[4][16 * 32];

  int wave = threadIdx.x >> 5;
  int lane = threadIdx.x & 31;
  int wid = blockIdx.x * 4 + wave;
  int qt = wid & (kT / 16 - 1);          // 0..127
  int hh = (wid >> 7) & (kH - 1);
  int b  = wid >> 11;
  int q0 = qt * 16;

  const h16* qbase = qs + ((size_t)(b * kH + hh) * kT + q0) * kDH;
  const h16* kbase = ks + (size_t)(b * kH + hh) * kT * kDH;
  const h16* vbase = vt + (size_t)(b * kH + hh) * kDH * kT;

  v16h qa0 = load_a_frag(qbase, kDH);
  v16h qa1 = load_a_frag(qbase + 32, kDH);

  v8f acc[4];
#pragma unroll
  for (int f = 0; f < 4; ++f) acc[f] = v8f{};
  float Mrow[8], Lrow[8];
#pragma unroll
  for (int j = 0; j < 8; ++j) { Mrow[j] = -__builtin_inff(); Lrow[j] = 0.f; }

  int hs = lane >> 4, n16 = lane & 15;
  _Float16* pl = &Pb[wave][0];
  int ktiles = (q0 + 16 + 31) >> 5;      // 32-wide key tiles up to diagonal

  for (int kt = 0; kt < ktiles; ++kt) {
    int j0 = kt * 32;
    v8f s[2];
#pragma unroll
    for (int nt = 0; nt < 2; ++nt) {
      s[nt] = v8f{};
      const h16* kb = kbase + (size_t)(j0 + 16 * nt) * kDH;
      s[nt] = wmma_f16(qa0, load_b_frag(kb, kDH), s[nt]);
      s[nt] = wmma_f16(qa1, load_b_frag(kb + 32, kDH), s[nt]);
    }
#pragma unroll
    for (int nt = 0; nt < 2; ++nt)
#pragma unroll
      for (int j = 0; j < 8; ++j) {
        int irow = q0 + j + 8 * hs;
        int col  = j0 + nt * 16 + n16;
        float v = s[nt][j] * 0.125f;     // 1/sqrt(64)
        s[nt][j] = (col > irow) ? -1.0e30f : v;
      }
    float nM[8], cf[8];
#pragma unroll
    for (int j = 0; j < 8; ++j) {
      float rm = fmaxf(s[0][j], s[1][j]);
      rm = fmaxf(rm, __shfl_xor(rm, 1, 32));
      rm = fmaxf(rm, __shfl_xor(rm, 2, 32));
      rm = fmaxf(rm, __shfl_xor(rm, 4, 32));
      rm = fmaxf(rm, __shfl_xor(rm, 8, 32));
      nM[j] = fmaxf(Mrow[j], rm);
      cf[j] = __expf(Mrow[j] - nM[j]);
      Mrow[j] = nM[j];
    }
#pragma unroll
    for (int f = 0; f < 4; ++f)
#pragma unroll
      for (int j = 0; j < 8; ++j) acc[f][j] *= cf[j];
#pragma unroll
    for (int j = 0; j < 8; ++j) {
      float p0 = __expf(s[0][j] - nM[j]);
      float p1 = __expf(s[1][j] - nM[j]);
      s[0][j] = p0; s[1][j] = p1;
      float r = p0 + p1;
      r += __shfl_xor(r, 1, 32);
      r += __shfl_xor(r, 2, 32);
      r += __shfl_xor(r, 4, 32);
      r += __shfl_xor(r, 8, 32);
      Lrow[j] = Lrow[j] * cf[j] + r;
    }
#pragma unroll
    for (int nt = 0; nt < 2; ++nt)
#pragma unroll
      for (int j = 0; j < 8; ++j) {
        int m = j + 8 * hs, jj = nt * 16 + n16;
        pl[m * 32 + jj] = (_Float16)s[nt][j];
      }
    asm volatile("s_wait_dscnt 0" ::: "memory");  // cross-lane LDS dependency
    v16h pa = load_a_frag(pl, 32);
#pragma unroll
    for (int f = 0; f < 4; ++f) {
      v16h vb = load_b_frag(vbase + (size_t)(f * 16) * kT + j0, kT);
      acc[f] = wmma_f16(pa, vb, acc[f]);
    }
  }

#pragma unroll
  for (int f = 0; f < 4; ++f)
#pragma unroll
    for (int j = 0; j < 8; ++j) {
      int m = j + 8 * hs;
      float v = acc[f][j] / Lrow[j];
      ctx[(size_t)(b * kT + q0 + m) * kD + hh * kDH + f * 16 + n16] = (h16)v;
    }
}

// ---------------------------------------------------------------------------
// LayerNorm over last dim (D=1024), one block (256 threads) per row.
// ---------------------------------------------------------------------------
__global__ __launch_bounds__(256) void layernorm_k(
    const float* __restrict__ in, const float* __restrict__ g,
    const float* __restrict__ bta, float* __restrict__ outF,
    h16* __restrict__ outH) {
  __shared__ float red[256];
  int row = blockIdx.x;
  int tid = threadIdx.x;
  const float* x = in + (size_t)row * kD;

  float s = 0.f;
  for (int i = tid; i < kD; i += 256) s += x[i];
  red[tid] = s; __syncthreads();
  for (int off = 128; off > 0; off >>= 1) {
    if (tid < off) red[tid] += red[tid + off];
    __syncthreads();
  }
  float mu = red[0] / kD;
  __syncthreads();

  float s2 = 0.f;
  for (int i = tid; i < kD; i += 256) { float d = x[i] - mu; s2 += d * d; }
  red[tid] = s2; __syncthreads();
  for (int off = 128; off > 0; off >>= 1) {
    if (tid < off) red[tid] += red[tid + off];
    __syncthreads();
  }
  float rstd = rsqrtf(red[0] / kD + 1e-5f);

  for (int i = tid; i < kD; i += 256) {
    float y = (x[i] - mu) * rstd * g[i] + bta[i];
    outF[(size_t)row * kD + i] = y;
    if (outH) outH[(size_t)row * kD + i] = (h16)y;
  }
}

// ---------------------------------------------------------------------------
// Host-side launch
// ---------------------------------------------------------------------------
extern "C" void kernel_launch(void* const* d_in, const int* in_sizes, int n_in,
                              void* d_out, int out_size, void* d_ws,
                              size_t ws_size, hipStream_t stream) {
  const float* x    = (const float*)d_in[0];
  const float* Wq   = (const float*)d_in[2];
  const float* bq   = (const float*)d_in[3];
  const float* Wk   = (const float*)d_in[4];
  const float* bk   = (const float*)d_in[5];
  const float* Wv   = (const float*)d_in[6];
  const float* bv   = (const float*)d_in[7];
  const float* Wo   = (const float*)d_in[8];
  const float* bo   = (const float*)d_in[9];
  const float* W1   = (const float*)d_in[10];
  const float* b1   = (const float*)d_in[11];
  const float* W2   = (const float*)d_in[12];
  const float* b2   = (const float*)d_in[13];
  const float* ln1g = (const float*)d_in[14];
  const float* ln1b = (const float*)d_in[15];
  const float* ln2g = (const float*)d_in[16];
  const float* ln2b = (const float*)d_in[17];
  float* out = (float*)d_out;

  char* ws = (char*)d_ws;
  size_t off = 0;
  auto alloc = [&](size_t bytes) -> void* {
    void* p = ws + off;
    off += (bytes + 255) & ~(size_t)255;
    return p;
  };
  h16* xh    = (h16*)alloc((size_t)kM * kD * 2);
  h16* wqh   = (h16*)alloc((size_t)kD * kD * 2);
  h16* wkh   = (h16*)alloc((size_t)kD * kD * 2);
  h16* wvh   = (h16*)alloc((size_t)kD * kD * 2);
  h16* woh   = (h16*)alloc((size_t)kD * kD * 2);
  h16* w1h   = (h16*)alloc((size_t)kF * kD * 2);
  h16* w2h   = (h16*)alloc((size_t)kD * kF * 2);
  h16* qs    = (h16*)alloc((size_t)kM * kD * 2);
  h16* ks    = (h16*)alloc((size_t)kM * kD * 2);
  h16* vts   = (h16*)alloc((size_t)kM * kD * 2);
  h16* ctxh  = (h16*)alloc((size_t)kM * kD * 2);
  h16* ffn1h = (h16*)alloc((size_t)kM * kF * 2);
  h16* x1h   = (h16*)alloc((size_t)kM * kD * 2);
  float* attn_res = (float*)alloc((size_t)kM * kD * 4);
  float* x1       = (float*)alloc((size_t)kM * kD * 4);
  float* y2       = (float*)alloc((size_t)kM * kD * 4);

  auto cvt = [&](const float* src, h16* dst, int n) {
    cvt_f32_f16<<<(n + 255) / 256, 256, 0, stream>>>(src, dst, n);
  };
  cvt(x, xh, kM * kD);
  cvt(Wq, wqh, kD * kD);
  cvt(Wk, wkh, kD * kD);
  cvt(Wv, wvh, kD * kD);
  cvt(Wo, woh, kD * kD);
  cvt(W1, w1h, kF * kD);
  cvt(W2, w2h, kD * kF);

  // Q, K (B,H,T,dh) and V transposed (B,H,dh,T)
  dim3 gP(kD / 64, kM / 128);
  gemm_ws<<<gP, 128, 0, stream>>>(xh, wqh, bq, nullptr, nullptr, qs,
                                  kM, kD, kD, /*mode*/1, 0);
  gemm_ws<<<gP, 128, 0, stream>>>(xh, wkh, bk, nullptr, nullptr, ks,
                                  kM, kD, kD, 1, 0);
  gemm_ws<<<gP, 128, 0, stream>>>(xh, wvh, bv, nullptr, nullptr, vts,
                                  kM, kD, kD, 2, 0);

  // causal flash attention -> ctx (B,T,D) f16
  attn_kernel<<<(kB * kH * (kT / 16)) / 4, 128, 0, stream>>>(qs, ks, vts, ctxh);

  // out-projection + residual(x) -> attn_res (f32)
  gemm_ws<<<gP, 128, 0, stream>>>(ctxh, woh, bo, x, attn_res, nullptr,
                                  kM, kD, kD, 3, 0);
  // LN1 -> x1 (f32) + x1h (f16)
  layernorm_k<<<kM, 256, 0, stream>>>(attn_res, ln1g, ln1b, x1, x1h);

  // FFN1 (ReLU) -> ffn1h (f16)
  dim3 gF(kF / 64, kM / 128);
  gemm_ws<<<gF, 128, 0, stream>>>(x1h, w1h, b1, nullptr, nullptr, ffn1h,
                                  kM, kF, kD, 0, 1);
  // FFN2 + residual(x1) -> y2 (f32)
  gemm_ws<<<gP, 128, 0, stream>>>(ffn1h, w2h, b2, x1, y2, nullptr,
                                  kM, kD, kF, 3, 0);
  // LN2 -> final output (f32)
  layernorm_k<<<kM, 256, 0, stream>>>(y2, ln2g, ln2b, out, nullptr);
}
